// CLUB_816043786555
// MI455X (gfx1250) — matrix operand order repaired and checked by reading, not compiled, not run on hardware
//
#include <hip/hip_runtime.h>
#include <hip/hip_bf16.h>

typedef _Float16 h16 __attribute__((ext_vector_type(16)));
typedef _Float16 h8  __attribute__((ext_vector_type(8)));
typedef float    f8  __attribute__((ext_vector_type(8)));
typedef float    f4  __attribute__((ext_vector_type(4)));

static constexpr int kN  = 131072;
static constexpr int kX  = 128;
static constexpr int kH2 = 512;
static constexpr int kY  = 128;
static constexpr int kWaves = 8;   // waves per block; phase-aligned for WGP$ weight reuse

// ---------------- init: zero accumulators in ws ----------------
__global__ void club_init(float* __restrict__ ym, float* __restrict__ y2m,
                          float* __restrict__ acc) {
  int t = threadIdx.x;
  if (t < kY) { ym[t] = 0.f; y2m[t] = 0.f; }
  if (t == 0) acc[0] = 0.f;
}

// ---------------- y statistics: E[y], E[y^2] per dim ----------------
__global__ void club_ystats(const float* __restrict__ y, float* __restrict__ ym,
                            float* __restrict__ y2m) {
  const int d   = threadIdx.x & (kY - 1);
  const int sub = threadIdx.x >> 7;      // 0..1
  const int r0  = blockIdx.x * 512;
  float s = 0.f, s2 = 0.f;
  for (int r = r0 + sub; r < r0 + 512; r += 2) {
    float v = y[(size_t)r * kY + d];
    s += v; s2 += v * v;
  }
  const float inv = 1.0f / (float)kN;
  atomicAdd(&ym[d],  s  * inv);
  atomicAdd(&y2m[d], s2 * inv);
}

// ---------------- convert weights f32 -> f16 into ws ----------------
// ws layout (halves): [W1m:65536][W1v:65536][W2m:65536][W2v:65536]
__global__ void club_convert(const float* __restrict__ W1m, const float* __restrict__ W1v,
                             const float* __restrict__ W2m, const float* __restrict__ W2v,
                             _Float16* __restrict__ out) {
  int idx = blockIdx.x * blockDim.x + threadIdx.x;   // < 262144
  int seg = idx >> 16, off = idx & 65535;
  float v = (seg == 0) ? W1m[off] : (seg == 1) ? W1v[off]
          : (seg == 2) ? W2m[off] : W2v[off];
  out[idx] = (_Float16)v;
}

// ---------------- main fused MLP + estimator ----------------
__global__ __launch_bounds__(kWaves * 32) void club_main(
    const float* __restrict__ x, const float* __restrict__ y,
    const _Float16* __restrict__ Wh,   // [W1m|W1v|W2m|W2v] in f16
    const float* __restrict__ b1m, const float* __restrict__ b2m,
    const float* __restrict__ b1v, const float* __restrict__ b2v,
    const float* __restrict__ ym, const float* __restrict__ y2m,
    float* __restrict__ acc)
{
  // per-wave staging of one 16 x 128 hidden chunk (f16): 4 KB/wave, 32 KB/block
  __shared__ _Float16 Hbuf[kWaves][16 * 128];

  const int wave = threadIdx.x >> 5;
  const int lane = threadIdx.x & 31;
  const int hi16 = lane >> 4;
  const int l15  = lane & 15;
  const int r0   = (blockIdx.x * kWaves + wave) * 16;

  _Float16* __restrict__ H = Hbuf[wave];

  // ---- A fragments of x (f32 -> f16); ISA A-layout for 16x16x32 f16:
  // lanes 0-15: row M=l15, K = {c*32+0..7, c*32+16..23}
  // lanes16-31: row M=l15, K = {c*32+8..15, c*32+24..31}
  h16 ax[4];
  {
    const float* xrow = x + (size_t)(r0 + l15) * kX;
    #pragma unroll
    for (int c = 0; c < 4; ++c) {
      const int ka = c * 32 + hi16 * 8;
      f4 a0 = *(const f4*)(xrow + ka);
      f4 a1 = *(const f4*)(xrow + ka + 4);
      f4 a2 = *(const f4*)(xrow + ka + 16);
      f4 a3 = *(const f4*)(xrow + ka + 20);
      h16 a;
      #pragma unroll
      for (int e = 0; e < 4; ++e) {
        a[e]      = (_Float16)a0[e];
        a[e + 4]  = (_Float16)a1[e];
        a[e + 8]  = (_Float16)a2[e];
        a[e + 12] = (_Float16)a3[e];
      }
      ax[c] = a;
    }
  }

  const f8 fz = {};
  f8 oaccm[8], oaccv[8];
  #pragma unroll
  for (int t = 0; t < 8; ++t) { oaccm[t] = fz; oaccv[t] = fz; }

  // One branch: out[16x128] += sum over 4 hidden chunks of leaky(x@W1+b1) @ W2
  auto run_branch = [&](const _Float16* __restrict__ W1,
                        const _Float16* __restrict__ W2,
                        const float* __restrict__ b1,
                        f8 (&oacc)[8]) {
    #pragma unroll 1
    for (int cb = 0; cb < 4; ++cb) {          // 128 hidden units per chunk
      // GEMM1: H[16 x 128] = leaky(x @ W1[:, cb*128 .. +128) + b1)
      #pragma unroll 1
      for (int nt = 0; nt < 8; ++nt) {
        f8 hacc = fz;
        #pragma unroll
        for (int c = 0; c < 4; ++c) {
          // B layout: lane column N = l15, K in table order (c*32 + 8*hi16 + {0..7, 16..23})
          h16 b;
          #pragma unroll
          for (int s = 0; s < 16; ++s)
            b[s] = W1[(size_t)(c * 32 + hi16 * 8 + (s & 7) + (s >> 3) * 16) * kH2 + cb * 128 + nt * 16 + l15];
          hacc = __builtin_amdgcn_wmma_f32_16x16x32_f16(false, ax[c], false, b,
                                                        (short)0, hacc, false, false);
        }
        const float bias = b1[cb * 128 + nt * 16 + l15];
        #pragma unroll
        for (int j = 0; j < 8; ++j) {
          // D layout: VGPR j -> row M = j + 8*hi16, col N = l15
          float h = hacc[j] + bias;
          h = (h >= 0.f) ? h : 0.2f * h;
          H[(j + 8 * hi16) * 128 + nt * 16 + l15] = (_Float16)h;
        }
      }
      __builtin_amdgcn_wave_barrier();   // same-wave LDS is in-order; keep scheduler honest

      // GEMM2 partial: oacc += H @ W2[cb*128 .. +128, :]
      #pragma unroll 1
      for (int c = 0; c < 4; ++c) {
        const int ka = c * 32 + hi16 * 8;
        h8 lo = *(const h8*)(H + l15 * 128 + ka);
        h8 hh = *(const h8*)(H + l15 * 128 + ka + 16);
        h16 a = __builtin_shufflevector(lo, hh, 0,1,2,3,4,5,6,7,8,9,10,11,12,13,14,15);
        #pragma unroll
        for (int nt = 0; nt < 8; ++nt) {
          h16 b;
          #pragma unroll
          for (int s = 0; s < 16; ++s)
            b[s] = W2[(size_t)(cb * 128 + c * 32 + hi16 * 8 + (s & 7) + (s >> 3) * 16) * kY + nt * 16 + l15];
          oacc[nt] = __builtin_amdgcn_wmma_f32_16x16x32_f16(false, a, false, b,
                                                            (short)0, oacc[nt], false, false);
        }
      }
      __builtin_amdgcn_wave_barrier();
      // keep all 8 waves phase-aligned: their weight loads for the next chunk
      // coincide in time, so the 32 KB chunk working set hits in the WGP$
      __syncthreads();
    }
  };

  run_branch(Wh,          Wh + 131072, b1m, oaccm);   // mu branch
  run_branch(Wh + 65536,  Wh + 196608, b1v, oaccv);   // logvar branch

  // fused estimator, all in registers:
  // 0.5*exp(-tanh(ov)) * ((y2m - 2 mu ym + mu^2) - (mu - y)^2)
  float partial = 0.f;
  #pragma unroll
  for (int nt = 0; nt < 8; ++nt) {
    const int d = nt * 16 + l15;
    const float bm = b2m[d], bv = b2v[d];
    const float yme = ym[d], y2me = y2m[d];
    #pragma unroll
    for (int j = 0; j < 8; ++j) {
      const int m = j + 8 * hi16;
      float mu  = oaccm[nt][j] + bm;
      float lv  = tanhf(oaccv[nt][j] + bv);
      float iv  = __expf(-lv);
      float yv  = y[(size_t)(r0 + m) * kY + d];
      float dmy = mu - yv;
      float nsq = y2me - 2.f * mu * yme + mu * mu;
      partial += 0.5f * iv * (nsq - dmy * dmy);
    }
  }

  // wave32 reduction then one atomic per wave
  #pragma unroll
  for (int off = 16; off > 0; off >>= 1)
    partial += __shfl_xor(partial, off, 32);
  if (lane == 0) atomicAdd(acc, partial);
}

__global__ void club_finalize(const float* __restrict__ acc, float* __restrict__ out) {
  out[0] = acc[0] * (1.0f / (float)kN);
}

extern "C" void kernel_launch(void* const* d_in, const int* in_sizes, int n_in,
                              void* d_out, int out_size, void* d_ws, size_t ws_size,
                              hipStream_t stream) {
  const float* x   = (const float*)d_in[0];
  const float* y   = (const float*)d_in[1];
  const float* W1m = (const float*)d_in[2];
  const float* b1m = (const float*)d_in[3];
  const float* W2m = (const float*)d_in[4];
  const float* b2m = (const float*)d_in[5];
  const float* W1v = (const float*)d_in[6];
  const float* b1v = (const float*)d_in[7];
  const float* W2v = (const float*)d_in[8];
  const float* b2v = (const float*)d_in[9];
  float* out = (float*)d_out;

  _Float16* Wh = (_Float16*)d_ws;                    // 262144 halves = 512 KB
  float* ym  = (float*)((char*)d_ws + 524288);       // 128 f32
  float* y2m = ym + kY;                              // 128 f32
  float* acc = y2m + kY;                             // 1 f32

  club_init<<<1, 256, 0, stream>>>(ym, y2m, acc);
  club_ystats<<<kN / 512, 256, 0, stream>>>(y, ym, y2m);
  club_convert<<<262144 / 256, 256, 0, stream>>>(W1m, W1v, W2m, W2v, Wh);
  club_main<<<kN / (kWaves * 16), kWaves * 32, 0, stream>>>(
      x, y, Wh, b1m, b2m, b1v, b2v, ym, y2m, acc);
  club_finalize<<<1, 1, 0, stream>>>(acc, out);
}
